// RFPrismModule_44324062494747
// MI455X (gfx1250) — matrix operand order repaired and checked by reading, not compiled, not run on hardware
//
#include <hip/hip_runtime.h>
#include <hip/hip_bf16.h>

// ---------------------------------------------------------------------------
// RF Prism per-subcarrier MLP, fused 3 layers, f16 WMMA with f32 accumulate.
// Grid: (BATCH/BM, C). Block: 256 threads = 8 wave32 waves.
// Software-pipelined fragment feed (B double-buffered x2, A prefetched early).
// ---------------------------------------------------------------------------

#define C_DIM   256
#define I_DIM   256
#define H_DIM   256
#define BATCH_N 2048
#define BM      128          // batch rows per workgroup
#define LDW     264          // padded LDS row stride in halves (264*2B = 528B, 16B-aligned)

typedef __attribute__((ext_vector_type(16))) _Float16 v16h;
typedef __attribute__((ext_vector_type(8)))  _Float16 v8h;
typedef __attribute__((ext_vector_type(8)))  float    v8f;

__device__ __forceinline__ v16h frag_cat(v8h lo, v8h hi) {
    return __builtin_shufflevector(lo, hi, 0,1,2,3,4,5,6,7,8,9,10,11,12,13,14,15);
}

// A-fragment (16x32 f16, MxK): lane L<16 holds row M=L, K=[b..b+7] & [b+16..b+23];
// lane L+16 holds row M=L, K=[b+8..b+15] & [b+24..b+31]. smem row-major [row][K].
__device__ __forceinline__ v16h load_a_frag(const _Float16* sA, int row, int kbase,
                                            int hi /*lane>>4*/) {
    const _Float16* p = sA + row * LDW + kbase + hi * 8;
    v8h lo  = *(const v8h*)p;
    v8h hi8 = *(const v8h*)(p + 16);
    return frag_cat(lo, hi8);
}

// B-fragment (32x16 f16, KxN): lane L<16 holds col N=L, K=[b..b+7] & [b+16..b+23];
// lane L+16 holds col N=L, K=[b+8..b+15] & [b+24..b+31]. smem stores W transposed: sW[n][k].
__device__ __forceinline__ v16h load_b_frag(const _Float16* sW, int ncol, int kbase,
                                            int hi /*lane>>4*/) {
    const _Float16* p = sW + ncol * LDW + kbase + hi * 8;
    v8h lo  = *(const v8h*)p;
    v8h hi8 = *(const v8h*)(p + 16);
    return frag_cat(lo, hi8);
}

// One 16x256 output stripe: acc[16] (16x16 f32 tiles) over K=256, pipelined.
__device__ __forceinline__ void gemm_stripe(const _Float16* sA, const _Float16* sW,
                                            int arow, int l15, int hi, v8f acc[16]) {
    v16h a_cur = load_a_frag(sA, arow, 0, hi);
    v16h a_nxt = a_cur;
    v16h b0 = load_b_frag(sW, l15, 0, hi);
    v16h b1 = load_b_frag(sW, 16 + l15, 0, hi);
    #pragma unroll
    for (int i = 0; i < 128; ++i) {
        const int nt = i & 15;
        const int ks = i >> 4;
        // Prefetch B fragment two steps ahead.
        v16h b2 = b1;
        if (i + 2 < 128) {
            const int j = i + 2;
            b2 = load_b_frag(sW, (j & 15) * 16 + l15, (j >> 4) * 32, hi);
        }
        // Prefetch next k-step's A fragment a few steps early.
        if (nt == 13 && ks < 7) a_nxt = load_a_frag(sA, arow, (ks + 1) * 32, hi);
        acc[nt] = __builtin_amdgcn_wmma_f32_16x16x32_f16(
            false, a_cur, false, b0, (short)0, acc[nt], false, false);
        b0 = b1; b1 = b2;
        if (nt == 15) a_cur = a_nxt;
    }
}

__global__ void __launch_bounds__(256, 1)
rf_prism_fused_kernel(const float* __restrict__ x,    // [B, I]
                      const float* __restrict__ W1,   // [C, I, H]
                      const float* __restrict__ b1,   // [C, H]
                      const float* __restrict__ W2,   // [C, H, H]
                      const float* __restrict__ b2,   // [C, H]
                      const float* __restrict__ W3,   // [C, H]
                      const float* __restrict__ b3,   // [C]
                      float* __restrict__ out)        // [B, C]
{
    __shared__ _Float16 sW[H_DIM * LDW];   // weights, transposed: sW[n*LDW + k]  (132 KB)
    __shared__ _Float16 sA[BM * LDW];      // activations row-major: sA[r*LDW + k] (66 KB)

    const int tid   = threadIdx.x;
    const int tileM = blockIdx.x;          // batch tile (fast-varying -> L2 reuse of W[c])
    const int c     = blockIdx.y;          // subcarrier
    const int wave  = tid >> 5;
    const int lane  = tid & 31;
    const int l15   = lane & 15;
    const int hi    = lane >> 4;

    const float* W1c = W1 + (size_t)c * (I_DIM * H_DIM);
    const float* W2c = W2 + (size_t)c * (H_DIM * H_DIM);

    // ---- Stage x tile (f32 -> f16, row-major, packed stores) ----
    {
        const float2* xg = (const float2*)(x + (size_t)tileM * BM * I_DIM);
        #pragma unroll
        for (int i = 0; i < (BM * I_DIM) / (256 * 2); ++i) {
            int i2 = tid + i * 256;                 // float2 index
            float2 v = xg[i2];
            int idx = i2 * 2;
            int r = idx >> 8, k = idx & 255;        // k even -> 4B-aligned pair store
            union { _Float16 h[2]; unsigned u; } pk;
            pk.h[0] = (_Float16)v.x;
            pk.h[1] = (_Float16)v.y;
            *(unsigned*)&sA[r * LDW + k] = pk.u;
        }
    }
    // ---- Stage W1[c] (f32 -> f16, transposed) ----
    {
        const float2* wg = (const float2*)W1c;
        #pragma unroll
        for (int i = 0; i < (I_DIM * H_DIM) / (256 * 2); ++i) {
            int i2 = tid + i * 256;
            float2 v = wg[i2];
            int idx = i2 * 2;
            int k = idx >> 8, n = idx & 255;        // W1c[idx] = W1[c][k][n]
            sW[n * LDW + k]       = (_Float16)v.x;
            sW[(n + 1) * LDW + k] = (_Float16)v.y;
        }
    }
    __syncthreads();

    // Prefetch W2[c] (256 KB = 2048 cachelines) while layer-1 math runs.
    {
        const char* pf = (const char*)W2c;
        #pragma unroll
        for (int i = 0; i < 8; ++i)
            __builtin_prefetch(pf + (tid + i * 256) * 128, 0, 1);
    }

    // =========================== Layer 1 ===========================
    v8f acc[16];
    #pragma unroll
    for (int nt = 0; nt < 16; ++nt) acc[nt] = (v8f)0.0f;

    const int arow = wave * 16 + l15;
    gemm_stripe(sA, sW, arow, l15, hi, acc);

    // Epilogue: bias + ReLU -> h1 (f16) written over this wave's own rows of sA.
    {
        const float* b1c = b1 + (size_t)c * H_DIM;
        #pragma unroll
        for (int nt = 0; nt < 16; ++nt) {
            int ncol = nt * 16 + l15;
            float bias = b1c[ncol];
            #pragma unroll
            for (int v = 0; v < 8; ++v) {
                float val = acc[nt][v] + bias;       // D layout: lane<16 -> M=v, else M=v+8
                val = fmaxf(val, 0.0f);
                int row = wave * 16 + v + hi * 8;
                sA[row * LDW + ncol] = (_Float16)val;
            }
        }
    }
    __syncthreads();

    // ---- Stage W2[c] (transposed f16) ----
    {
        const float2* wg = (const float2*)W2c;
        #pragma unroll
        for (int i = 0; i < (H_DIM * H_DIM) / (256 * 2); ++i) {
            int i2 = tid + i * 256;
            float2 v = wg[i2];
            int idx = i2 * 2;
            int k = idx >> 8, n = idx & 255;
            sW[n * LDW + k]       = (_Float16)v.x;
            sW[(n + 1) * LDW + k] = (_Float16)v.y;
        }
    }
    __syncthreads();

    // =========================== Layer 2 ===========================
    #pragma unroll
    for (int nt = 0; nt < 16; ++nt) acc[nt] = (v8f)0.0f;
    gemm_stripe(sA, sW, arow, l15, hi, acc);

    // ============== Layer 3: out = relu(h2) . W3[c] + b3[c] ==============
    {
        const float* b2c = b2 + (size_t)c * H_DIM;
        const float* W3c = W3 + (size_t)c * H_DIM;
        float part[8];
        #pragma unroll
        for (int v = 0; v < 8; ++v) part[v] = 0.0f;

        #pragma unroll
        for (int nt = 0; nt < 16; ++nt) {
            int ncol = nt * 16 + l15;
            float bias = b2c[ncol];
            float w3v  = W3c[ncol];
            #pragma unroll
            for (int v = 0; v < 8; ++v) {
                float val = fmaxf(acc[nt][v] + bias, 0.0f);
                part[v] += val * w3v;
            }
        }

        // Reduce across the 16 lanes of each half-wave (rows v / v+8).
        float b3c = b3[c];
        #pragma unroll
        for (int v = 0; v < 8; ++v) {
            float s = part[v];
            s += __shfl_xor(s, 1, 32);
            s += __shfl_xor(s, 2, 32);
            s += __shfl_xor(s, 4, 32);
            s += __shfl_xor(s, 8, 32);
            if (l15 == 0) {
                int row = tileM * BM + wave * 16 + v + hi * 8;
                out[(size_t)row * C_DIM + c] = s + b3c;
            }
        }
    }
}

extern "C" void kernel_launch(void* const* d_in, const int* in_sizes, int n_in,
                              void* d_out, int out_size, void* d_ws, size_t ws_size,
                              hipStream_t stream) {
    (void)in_sizes; (void)n_in; (void)d_ws; (void)ws_size; (void)out_size;
    const float* x  = (const float*)d_in[0];
    const float* W1 = (const float*)d_in[1];
    const float* b1 = (const float*)d_in[2];
    const float* W2 = (const float*)d_in[3];
    const float* b2 = (const float*)d_in[4];
    const float* W3 = (const float*)d_in[5];
    const float* b3 = (const float*)d_in[6];
    float* out = (float*)d_out;

    dim3 grid(BATCH_N / BM, C_DIM);  // (16, 256); x fast-varying -> W[c] stays hot in L2
    dim3 block(256);
    rf_prism_fused_kernel<<<grid, block, 0, stream>>>(x, W1, b1, W2, b2, W3, b3, out);
}
